// DistanceEdgeSelfCond_73392401154175
// MI455X (gfx1250) — matrix operand order repaired and checked by reading, not compiled, not run on hardware
//
#include <hip/hip_runtime.h>
#include <hip/hip_bf16.h>

typedef __attribute__((ext_vector_type(16))) _Float16 v16h;
typedef __attribute__((ext_vector_type(8)))  float    v8f;

#define NUM_RBF 32
#define D_EDGE  64
#define NATOMS  512
#define CUTOFF  10.0f

// One block per (b, n); 8 wave32s; each wave covers 4 m-tiles of 16.
// Per (m-tile, d-tile): one v_wmma_f32_16x16x32_f16 (K=32 == NUM_RBF).
// Output (537 MB, write-once) is stored non-temporally to avoid L2 thrash.
__global__ __launch_bounds__(256) void edge_rbf_wmma_kernel(
    const float* __restrict__ coords,  // [B, N, 3]
    const float* __restrict__ mask,    // [B, N]
    const float* __restrict__ W,       // [D_EDGE, NUM_RBF] row-major
    const float* __restrict__ bias,    // [D_EDGE]
    float* __restrict__ out)           // [B, N, N, D_EDGE]
{
  const int N    = NATOMS;
  const int lane = threadIdx.x & 31;
  const int wave = threadIdx.x >> 5;
  const int col  = lane & 15;              // WMMA N index / A row index
  const bool hi  = (lane & 16) != 0;

  const int bn = blockIdx.x;               // b*N + n
  const int b  = bn >> 9;                  // / 512

  // Gaussian smearing constants: centers k*spacing, coeff = -0.5/spacing^2
  const float spacing = CUTOFF / (float)(NUM_RBF - 1);
  const float coeff   = -0.5f / (spacing * spacing);

  // ---- Preload B tiles (W^T as 32x16 f16) into registers.
  // Contiguous-k B order used here: lane = column (d within tile); lanes 0-15 hold K=0..15,
  // lanes 16-31 hold K=16..31; VGPR v packs K = kbase+2v, kbase+2v+1.
  const int kbase = hi ? 16 : 0;
  v16h Bt[4];
  float bb[4];
#pragma unroll
  for (int t = 0; t < 4; ++t) {
    const float* wrow = W + (size_t)(t * 16 + col) * NUM_RBF + kbase;
#pragma unroll
    for (int j = 0; j < 16; ++j) Bt[t][j] = (_Float16)wrow[j];
    bb[t] = bias[t * 16 + col];
  }

  // ---- Row atom (n) data: uniform across the block.
  const float xn = coords[(size_t)bn * 3 + 0];
  const float yn = coords[(size_t)bn * 3 + 1];
  const float zn = coords[(size_t)bn * 3 + 2];
  const float mn = mask[bn];

  // A (16x32) in the same contiguous-k order as B: lanes 0-15 K in {0..15},
  // lanes 16-31 K in {16..31}. Half h -> K = koff + h.
  const int koff = hi ? 16 : 0;

  for (int tile = wave; tile < N / 16; tile += 8) {
    const int mbase = tile * 16;
    const int m = mbase + col;             // this lane's A-matrix row atom

    const float dx = xn - coords[(size_t)(b * N + m) * 3 + 0];
    const float dy = yn - coords[(size_t)(b * N + m) * 3 + 1];
    const float dz = zn - coords[(size_t)(b * N + m) * 3 + 2];
    const float d2 = dx * dx + dy * dy + dz * dz;
    const float dist = (d2 > 0.0f) ? __builtin_sqrtf(d2) : 0.0f;

    // Build A: 16 RBF values per lane (16 exps), packed f16.
    v16h A;
#pragma unroll
    for (int h = 0; h < 16; ++h) {
      const int k = koff + h;
      const float diff = dist - (float)k * spacing;
      A[h] = (_Float16)__expf(coeff * diff * diff);
    }

    // 4 WMMAs: same A, four 16-wide d-tiles of W^T. K=32 in a single op.
    v8f acc[4];
#pragma unroll
    for (int t = 0; t < 4; ++t) {
      v8f c = {};
      acc[t] = __builtin_amdgcn_wmma_f32_16x16x32_f16(
          /*neg_a=*/false, A, /*neg_b=*/false, Bt[t],
          /*c_mod=*/(short)0, c, /*reuse_a=*/false, /*reuse_b=*/false);
    }

    // Epilogue: D-layout VGPR j holds M = j + (hi?8:0), N = col.
    // out[b][n][mbase+M][t*16+col] = (acc + bias) * mask_m * mask_n
    // Non-temporal: output stream is write-once, 3x larger than L2.
    const int mrow0 = mbase + (hi ? 8 : 0);
#pragma unroll
    for (int j = 0; j < 8; ++j) {
      const float scale = mask[b * N + mrow0 + j] * mn;
      float* orow = out + ((size_t)bn * N + (size_t)(mrow0 + j)) * D_EDGE + col;
#pragma unroll
      for (int t = 0; t < 4; ++t) {
        __builtin_nontemporal_store((acc[t][j] + bb[t]) * scale, orow + t * 16);
      }
    }
  }
}

extern "C" void kernel_launch(void* const* d_in, const int* in_sizes, int n_in,
                              void* d_out, int out_size, void* d_ws, size_t ws_size,
                              hipStream_t stream) {
  (void)in_sizes; (void)n_in; (void)out_size; (void)d_ws; (void)ws_size;
  const float* coords = (const float*)d_in[0];  // [8, 512, 3]
  const float* mask   = (const float*)d_in[1];  // [8, 512]
  const float* W      = (const float*)d_in[2];  // [64, 32]
  const float* bias   = (const float*)d_in[3];  // [64]
  float* out          = (float*)d_out;          // [8, 512, 512, 64]

  const int B = 8, N = NATOMS;
  dim3 grid(B * N);   // 4096 blocks, one per (b, n)
  dim3 block(256);    // 8 wave32s
  edge_rbf_wmma_kernel<<<grid, block, 0, stream>>>(coords, mask, W, bias, out);
}
